// MultiHAN_59536836657156
// MI455X (gfx1250) — matrix operand order repaired and checked by reading, not compiled, not run on hardware
//
#include <hip/hip_runtime.h>
#include <hip/hip_bf16.h>
#include <math.h>

#define NN 20000
#define CC 128
#define HH 8
#define DD 16
#define EE 400000
#define BB 32

typedef __attribute__((ext_vector_type(16))) __bf16 v16bf;
typedef __attribute__((ext_vector_type(8)))  float  v8f;

union BFrag { unsigned short u[16]; v16bf v; };

__device__ __forceinline__ unsigned short f2bf(float f) {
    unsigned int u = __float_as_uint(f);
    unsigned int r = u + 0x7FFFu + ((u >> 16) & 1u);   // round-to-nearest-even
    return (unsigned short)(r >> 16);
}

__device__ __forceinline__ void atomicMaxF(float* addr, float v) {
    if (v >= 0.0f) atomicMax((int*)addr, __float_as_int(v));
    else           atomicMin((unsigned int*)addr, (unsigned int)__float_as_int(v));
}

// branchless tanh: 1 - 2/(exp(2x)+1); exp overflow/underflow give +/-1 limits
__device__ __forceinline__ float tanh_fast(float x) {
    return 1.0f - 2.0f / (__expf(2.0f * x) + 1.0f);
}

// ---------------------------------------------------------------------------
// WMMA GEMM: Y[nrows,128] = X[nrows,128] @ W[128,128] + b
// mode 0: store Y.   mode 1: colsum[c] += sum_rows tanh(Y[r,c])  (Y not stored)
// Block: 256 thr = 8 waves, covers 64 rows x 128 cols. bf16 inputs, f32 acc.
// ---------------------------------------------------------------------------
__global__ __launch_bounds__(256) void gemm128_wmma(
    const float* __restrict__ X, const float* __restrict__ W,
    const float* __restrict__ bias, float* __restrict__ Y,
    float* __restrict__ colsum, int nrows, int mode)
{
    __shared__ unsigned short sA[64][128];   // 16 KB
    __shared__ unsigned short sB[128][128];  // 32 KB
    const int tid = threadIdx.x;
    const int row0 = blockIdx.x * 64;
    const bool full_tile = (row0 + 64 <= nrows);   // uniform

    #pragma unroll
    for (int i = 0; i < 64; ++i) {           // W: 16384 elems
        int idx = tid + i * 256;
        sB[idx >> 7][idx & 127] = f2bf(W[idx]);
    }
    #pragma unroll
    for (int i = 0; i < 32; ++i) {           // X tile: 8192 elems
        int idx = tid + i * 256;
        int r = idx >> 7, c = idx & 127;
        int gr = row0 + r;
        sA[r][c] = (gr < nrows) ? f2bf(X[(size_t)gr * CC + c]) : (unsigned short)0;
    }
    __syncthreads();

    const int lane = tid & 31, w = tid >> 5;
    const int msub  = (w & 3) * 16;          // row sub-tile
    const int nbase = (w >> 2) * 64;         // col group (0 or 64)
    const int half = lane >> 4, lm = lane & 15;

    // Preload A fragments for all 4 K-steps (ISA 16-bit A 16x32 layout)
    BFrag a[4];
    #pragma unroll
    for (int ks = 0; ks < 4; ++ks) {
        int k0 = ks * 32;
        #pragma unroll
        for (int e = 0; e < 16; ++e) {
            int K = k0 + (e & 7) + 16 * (e >> 3) + 8 * half;
            a[ks].u[e] = sA[msub + lm][K];
        }
    }

    for (int nt = 0; nt < 4; ++nt) {
        int n0 = nbase + nt * 16;
        float bval = bias[n0 + lm];
        v8f acc;
        #pragma unroll
        for (int r = 0; r < 8; ++r) acc[r] = bval;

        #pragma unroll
        for (int ks = 0; ks < 4; ++ks) {
            int k0 = ks * 32;
            BFrag bf;                         // ISA 16-bit B 32x16 layout
            #pragma unroll
            for (int e = 0; e < 16; ++e) {
                int K = k0 + (e & 7) + 16 * (e >> 3) + 8 * half;
                bf.u[e] = sB[K][n0 + lm];
            }
            acc = __builtin_amdgcn_wmma_f32_16x16x32_bf16(
                false, a[ks].v, false, bf.v, (short)0, acc, false, false);
        }

        if (mode == 0) {
            if (full_tile) {
                #pragma unroll
                for (int r = 0; r < 8; ++r) {
                    int gr = row0 + msub + r + 8 * half;
                    Y[(size_t)gr * CC + n0 + lm] = acc[r];
                }
            } else {
                #pragma unroll
                for (int r = 0; r < 8; ++r) {
                    int gr = row0 + msub + r + 8 * half;
                    if (gr < nrows) Y[(size_t)gr * CC + n0 + lm] = acc[r];
                }
            }
        } else {
            float s = 0.0f;
            #pragma unroll
            for (int r = 0; r < 8; ++r) {
                int gr = row0 + msub + r + 8 * half;
                float t = tanh_fast(acc[r]);          // branchless
                s += (gr < nrows) ? t : 0.0f;         // cndmask, no branch
            }
            atomicAdd(&colsum[n0 + lm], s);
        }
    }
}

// ---------------------------------------------------------------------------
// Small helper kernels
// ---------------------------------------------------------------------------
__global__ void fill_k(float* p, float v, int n) {
    int t = blockIdx.x * blockDim.x + threadIdx.x;
    if (t < n) p[t] = v;
}

// al[n,h] = sum_d h[n, h*16+d] * avec[h*16+d]
__global__ void compute_al(const float* __restrict__ hx,
                           const float* __restrict__ avec,
                           float* __restrict__ al, int nh)
{
    int t = blockIdx.x * blockDim.x + threadIdx.x;
    if (t >= nh) return;
    int n = t >> 3, h = t & 7;
    const float* hv = hx + (size_t)n * CC + h * DD;
    const float* av = avec + h * DD;
    float s = 0.0f;
    #pragma unroll
    for (int k = 0; k < DD; ++k) s += hv[k] * av[k];
    al[t] = s;
}

__device__ __forceinline__ float edge_score(const int* ei, const float* als,
                                            const float* ald, int e, int h,
                                            int* ps, int* pd)
{
    int s = ei[e], d = ei[EE + e];
    *ps = s; *pd = d;
    float sc = als[s * HH + h] + ald[d * HH + h];
    return sc > 0.0f ? sc : 0.2f * sc;     // leaky relu 0.2
}

__global__ void edge_max_k(const int* __restrict__ ei, const float* __restrict__ als,
                           const float* __restrict__ ald, float* __restrict__ m)
{
    int t = blockIdx.x * blockDim.x + threadIdx.x;
    if (t >= EE * HH) return;
    int e = t >> 3, h = t & 7, s, d;
    float sc = edge_score(ei, als, ald, e, h, &s, &d);
    atomicMaxF(&m[d * HH + h], sc);
}

__global__ void edge_expsum_k(const int* __restrict__ ei, const float* __restrict__ als,
                              const float* __restrict__ ald, const float* __restrict__ m,
                              float* __restrict__ z)
{
    int t = blockIdx.x * blockDim.x + threadIdx.x;
    if (t >= EE * HH) return;
    int e = t >> 3, h = t & 7, s, d;
    float sc = edge_score(ei, als, ald, e, h, &s, &d);
    atomicAdd(&z[d * HH + h], __expf(sc - m[d * HH + h]));
}

__global__ void edge_msg_k(const int* __restrict__ ei, const float* __restrict__ als,
                           const float* __restrict__ ald, const float* __restrict__ m,
                           const float* __restrict__ z, const float* __restrict__ hsrc,
                           float* __restrict__ o)
{
    int t = blockIdx.x * blockDim.x + threadIdx.x;
    if (t >= EE * HH) return;
    int e = t >> 3, h = t & 7, s, d;
    float sc = edge_score(ei, als, ald, e, h, &s, &d);
    float alpha = __expf(sc - m[d * HH + h]) / z[d * HH + h];
    const float* sv = hsrc + (size_t)s * CC + h * DD;
    float*       dv = o    + (size_t)d * CC + h * DD;
    #pragma unroll
    for (int k = 0; k < DD; ++k) atomicAdd(&dv[k], alpha * sv[k]);
}

__global__ void relu_k(float* p, int n) {
    int t = blockIdx.x * blockDim.x + threadIdx.x;
    if (t < n) p[t] = fmaxf(p[t], 0.0f);
}

// semantic attention: colsum[4][128] -> attn[4] (softmax over pairs {0,2},{1,3})
__global__ void semantic_k(const float* __restrict__ colsum,
                           const float* __restrict__ q, float* __restrict__ attn)
{
    if (threadIdx.x != 0) return;
    float sc[4];
    for (int j = 0; j < 4; ++j) {
        float s = 0.0f;
        for (int c = 0; c < CC; ++c) s += q[c] * colsum[j * CC + c];
        sc[j] = s / (float)NN;
    }
    {
        float mx = fmaxf(sc[0], sc[2]);
        float e0 = __expf(sc[0] - mx), e1 = __expf(sc[2] - mx);
        attn[0] = e0 / (e0 + e1); attn[1] = e1 / (e0 + e1);
    }
    {
        float mx = fmaxf(sc[1], sc[3]);
        float e0 = __expf(sc[1] - mx), e1 = __expf(sc[3] - mx);
        attn[2] = e0 / (e0 + e1); attn[3] = e1 / (e0 + e1);
    }
}

__global__ void combine_k(const float* __restrict__ oA, const float* __restrict__ oB,
                          const float* __restrict__ attn, int ai,
                          float* __restrict__ xout, int n)
{
    int t = blockIdx.x * blockDim.x + threadIdx.x;
    if (t >= n) return;
    float v = attn[ai] * oA[t] + attn[ai + 1] * oB[t];
    xout[t] = fmaxf(v, 0.0f);
}

__global__ void pool_atomic_k(const float* __restrict__ x, const int* __restrict__ batch,
                              float* __restrict__ fmax, float* __restrict__ fsum)
{
    int t = blockIdx.x * blockDim.x + threadIdx.x;
    if (t >= NN * CC) return;
    int n = t >> 7, c = t & 127;
    int b = batch[n];
    float v = x[t];
    atomicMaxF(&fmax[b * CC + c], v);
    atomicAdd(&fsum[b * CC + c], v);
}

__global__ void count_k(const int* __restrict__ batch, float* __restrict__ cnt) {
    int t = blockIdx.x * blockDim.x + threadIdx.x;
    if (t < NN) atomicAdd(&cnt[batch[t]], 1.0f);
}

__global__ void feat_fin_k(const float* __restrict__ fmax, const float* __restrict__ fsum,
                           const float* __restrict__ cnt, float* __restrict__ feat)
{
    int t = blockIdx.x * blockDim.x + threadIdx.x;
    if (t >= BB * 512) return;
    int b = t >> 9, c = t & 511;
    int ty = c >> 8, in_ = c & 255;
    float out;
    if (in_ < CC) {
        float v = fmax[ty * BB * CC + b * CC + in_];
        out = (isinf(v) && v < 0.0f) ? 0.0f : v;
    } else {
        float s = fsum[ty * BB * CC + b * CC + (in_ - CC)];
        out = s / fmaxf(cnt[ty * BB + b], 1.0f);
    }
    feat[b * 512 + c] = out;
}

__global__ void pairnorm_center_k(float* feat) {   // <<<1,512>>>
    int c = threadIdx.x;
    float s = 0.0f;
    for (int b = 0; b < BB; ++b) s += feat[b * 512 + c];
    float mean = s / (float)BB;
    for (int b = 0; b < BB; ++b) feat[b * 512 + c] -= mean;
}

__global__ void pairnorm_scale_k(float* feat) {    // <<<1,32>>>
    int b = threadIdx.x;
    float ss = 0.0f;
    for (int c = 0; c < 512; ++c) { float v = feat[b * 512 + c]; ss += v * v; }
    float inv = 100.0f / sqrtf(1e-6f + ss);
    for (int c = 0; c < 512; ++c) feat[b * 512 + c] *= inv;
}

__global__ void mlp_k(const float* __restrict__ X, const float* __restrict__ W,
                      const float* __restrict__ bias, float* __restrict__ Y,
                      int K, int Nc, int do_relu)
{
    int b = blockIdx.x, c = threadIdx.x;
    float s = bias[c];
    for (int k = 0; k < K; ++k) s += X[b * K + k] * W[k * Nc + c];
    if (do_relu) s = fmaxf(s, 0.0f);
    Y[b * Nc + c] = s;
}

// ---------------------------------------------------------------------------
extern "C" void kernel_launch(void* const* d_in, const int* in_sizes, int n_in,
                              void* d_out, int out_size, void* d_ws, size_t ws_size,
                              hipStream_t stream)
{
    const float* x_bold = (const float*)d_in[0];
    const float* x_dti  = (const float*)d_in[1];
    const float* proj_w = (const float*)d_in[2];
    const float* proj_b = (const float*)d_in[3];
    const float* a_src  = (const float*)d_in[4];
    const float* a_dst  = (const float*)d_in[5];
    const float* k_w    = (const float*)d_in[6];
    const float* k_b    = (const float*)d_in[7];
    const float* qv     = (const float*)d_in[8];
    const float* lin1_w = (const float*)d_in[9];
    const float* lin1_b = (const float*)d_in[10];
    const float* lin2_w = (const float*)d_in[11];
    const float* lin2_b = (const float*)d_in[12];
    const float* lin3_w = (const float*)d_in[13];
    const float* lin3_b = (const float*)d_in[14];
    const int* ei[4] = { (const int*)d_in[15], (const int*)d_in[16],
                         (const int*)d_in[17], (const int*)d_in[18] };
    const int* batch_bold = (const int*)d_in[19];
    const int* batch_dti  = (const int*)d_in[20];

    const int st[4] = {0, 0, 1, 1}, dt[4] = {0, 1, 0, 1};

    // workspace layout
    const size_t NC = (size_t)NN * CC;
    float* ws  = (float*)d_ws;
    float* xa0 = ws;                float* xa1 = xa0 + NC;
    float* h0  = xa1 + NC;          float* h1  = h0 + NC;
    float* o0  = h1 + NC;  float* o1 = o0 + NC;  float* o2 = o1 + NC;  float* o3 = o2 + NC;
    float* ob[4] = {o0, o1, o2, o3};
    float* als  = o3 + NC;          float* ald  = als + NN * HH;
    float* mbuf = ald + NN * HH;    float* zbuf = mbuf + NN * HH;
    float* colsum = zbuf + NN * HH; // 4*128
    float* attn = colsum + 4 * CC;  // 4
    float* fmax = attn + 4;         // 2*B*C
    float* fsum = fmax + 2 * BB * CC;
    float* cnt  = fsum + 2 * BB * CC;  // 2*B
    float* feat = cnt + 2 * BB;        // B*512
    float* mlp1 = feat + BB * 512;     // B*128
    float* mlp2 = mlp1 + BB * CC;      // B*64

    auto cdiv = [](int a, int b) { return (a + b - 1) / b; };
    const int gGemm = cdiv(NN, 64);
    const int gNH   = cdiv(NN * HH, 256);
    const int gNC   = cdiv((int)NC, 256);
    const int gEH   = cdiv(EE * HH, 256);
    const float NEG_INF = -__builtin_inff();

    const float* xin0 = x_bold;
    const float* xin1 = x_dti;

    for (int l = 0; l < 3; ++l) {
        // node projections (WMMA bf16, f32 acc)
        gemm128_wmma<<<gGemm, 256, 0, stream>>>(xin0, proj_w + (size_t)(l*2+0)*CC*CC,
                                                proj_b + (l*2+0)*CC, h0, colsum, NN, 0);
        gemm128_wmma<<<gGemm, 256, 0, stream>>>(xin1, proj_w + (size_t)(l*2+1)*CC*CC,
                                                proj_b + (l*2+1)*CC, h1, colsum, NN, 0);
        const float* hs[2] = {h0, h1};

        for (int e = 0; e < 4; ++e) {
            compute_al<<<gNH, 256, 0, stream>>>(hs[st[e]], a_src + (size_t)(l*4+e)*CC, als, NN*HH);
            compute_al<<<gNH, 256, 0, stream>>>(hs[dt[e]], a_dst + (size_t)(l*4+e)*CC, ald, NN*HH);
            fill_k<<<gNH, 256, 0, stream>>>(mbuf, NEG_INF, NN*HH);
            fill_k<<<gNH, 256, 0, stream>>>(zbuf, 0.0f, NN*HH);
            fill_k<<<gNC, 256, 0, stream>>>(ob[e], 0.0f, (int)NC);
            edge_max_k   <<<gEH, 256, 0, stream>>>(ei[e], als, ald, mbuf);
            edge_expsum_k<<<gEH, 256, 0, stream>>>(ei[e], als, ald, mbuf, zbuf);
            edge_msg_k   <<<gEH, 256, 0, stream>>>(ei[e], als, ald, mbuf, zbuf, hs[st[e]], ob[e]);
            relu_k<<<gNC, 256, 0, stream>>>(ob[e], (int)NC);
        }

        // semantic attention: fused tanh + column-sum via WMMA GEMM mode 1
        fill_k<<<2, 256, 0, stream>>>(colsum, 0.0f, 4 * CC);
        for (int e = 0; e < 4; ++e)
            gemm128_wmma<<<gGemm, 256, 0, stream>>>(ob[e], k_w + (size_t)l*CC*CC,
                                                    k_b + l*CC, nullptr,
                                                    colsum + e*CC, NN, 1);
        semantic_k<<<1, 32, 0, stream>>>(colsum, qv + l*CC, attn);

        combine_k<<<gNC, 256, 0, stream>>>(o0, o2, attn, 0, xa0, (int)NC); // type 0: {bb, db}
        combine_k<<<gNC, 256, 0, stream>>>(o1, o3, attn, 2, xa1, (int)NC); // type 1: {bd, dd}
        xin0 = xa0; xin1 = xa1;
    }

    // pooling
    fill_k<<<cdiv(2*BB*CC,256), 256, 0, stream>>>(fmax, NEG_INF, 2*BB*CC);
    fill_k<<<cdiv(2*BB*CC,256), 256, 0, stream>>>(fsum, 0.0f, 2*BB*CC);
    fill_k<<<1, 64, 0, stream>>>(cnt, 0.0f, 2*BB);
    pool_atomic_k<<<gNC, 256, 0, stream>>>(xa0, batch_bold, fmax, fsum);
    pool_atomic_k<<<gNC, 256, 0, stream>>>(xa1, batch_dti, fmax + BB*CC, fsum + BB*CC);
    count_k<<<cdiv(NN,256), 256, 0, stream>>>(batch_bold, cnt);
    count_k<<<cdiv(NN,256), 256, 0, stream>>>(batch_dti, cnt + BB);
    feat_fin_k<<<cdiv(BB*512,256), 256, 0, stream>>>(fmax, fsum, cnt, feat);

    // PairNorm (PN-SI, scale 100)
    pairnorm_center_k<<<1, 512, 0, stream>>>(feat);
    pairnorm_scale_k<<<1, 32, 0, stream>>>(feat);

    // MLP head
    mlp_k<<<BB, 128, 0, stream>>>(feat, lin1_w, lin1_b, mlp1, 512, 128, 1);
    mlp_k<<<BB, 64, 0, stream>>>(mlp1, lin2_w, lin2_b, mlp2, 128, 64, 1);
    mlp_k<<<BB, 2, 0, stream>>>(mlp2, lin3_w, lin3_b, (float*)d_out, 64, 2, 0);
}
